// MCAT_Decoder_41566693491448
// MI455X (gfx1250) — matrix operand order repaired and checked by reading, not compiled, not run on hardware
//
#include <hip/hip_runtime.h>

typedef __attribute__((ext_vector_type(16))) __bf16 v16bf;
typedef __attribute__((ext_vector_type(8)))  float  v8f;

#define WMMA_BF16(a,b,c) \
  __builtin_amdgcn_wmma_f32_16x16x32_bf16(false,(a),false,(b),(short)0,(c),false,false)

// Problem dims (fixed by the reference)
constexpr int Bb = 32;     // batch
constexpr int Ss = 1024;   // memory sequence
constexpr int F0 = 2048;   // input feature
constexpr int Dd = 768;    // d_model
constexpr int Ll = 100;    // queries
constexpr int Hh = 8;      // heads
constexpr int HDh = 96;    // head dim
constexpr int Cc = 1000;   // classes
constexpr int DFf = 10;    // duplicate factor

// ---------------------------------------------------------------------------
// bf16 WMMA GEMM:  C[M,N] = act( A[M,K] * W[N,K]^T + bias[N] )
//   - 128x64 block tile, 8 wave32s, each wave owns a 32x32 sub-tile
//     (2x2 fragments -> 4 v_wmma per K-step, A/B fragment reuse)
//   - double-buffered LDS: next tile's global loads issue before the current
//     tile's WMMAs; one barrier per K-step
//   - A is f32 (converted while staging) or bf16; W is always f32.
// ---------------------------------------------------------------------------
template<bool A_F32, bool RELU, bool OUT_BF16>
__global__ __launch_bounds__(256)
void gemm_bf16_wmma(const void* __restrict__ Aptr, const float* __restrict__ W,
                    const float* __restrict__ bias, float* __restrict__ Cf,
                    __bf16* __restrict__ Cb, int M, int N, int K)
{
  constexpr int BM = 128, BN = 64, BK = 32, LDP = BK + 8; // pad rows: 80B stride
  __shared__ __bf16 As[2][BM * LDP];   // 2 * 10240 B
  __shared__ __bf16 Ws[2][BN * LDP];   // 2 *  5120 B   (30 KB total)

  const int tid  = threadIdx.x;
  const int wv   = tid >> 5;        // wave 0..7
  const int lane = tid & 31;
  const int wm   = wv >> 1;         // 0..3  (32-row slab)
  const int wn   = wv & 1;          // 0..1  (32-col slab)
  const int lr    = lane & 15;
  const int khalf = lane >> 4;

  // staging: A tile 128x32 -> 16 elems/thread; W tile 64x32 -> 8 elems/thread
  const int arow = tid >> 1;           // 0..127
  const int acol = (tid & 1) * 16;     // 0,16
  const int wrow = tid >> 2;           // 0..63
  const int wcol = (tid & 3) * 8;      // 0,8,16,24
  int arI = blockIdx.y * BM + arow; if (arI >= M) arI = M - 1;   // clamp reads
  const size_t aRow = (size_t)arI;
  const size_t wRow = (size_t)(blockIdx.x * BN + wrow);          // N % 64 == 0

  const float*  Af = (const float*)Aptr;
  const __bf16* Ab = (const __bf16*)Aptr;

  __bf16 a_stage[16];
  __bf16 w_stage[8];

  auto load_stage = [&](int k0) {
#pragma unroll
    for (int j = 0; j < 16; ++j)
      a_stage[j] = A_F32 ? (__bf16)Af[aRow * K + k0 + acol + j]
                         : Ab[aRow * K + k0 + acol + j];
#pragma unroll
    for (int j = 0; j < 8; ++j)
      w_stage[j] = (__bf16)W[wRow * K + k0 + wcol + j];
  };
  auto store_stage = [&](int p) {
#pragma unroll
    for (int j = 0; j < 16; ++j) As[p][arow * LDP + acol + j] = a_stage[j];
#pragma unroll
    for (int j = 0; j < 8; ++j)  Ws[p][wrow * LDP + wcol + j] = w_stage[j];
  };

  v8f acc00 = {}, acc01 = {}, acc10 = {}, acc11 = {};

  const int nT = K / BK;
  load_stage(0);
  for (int it = 0; it < nT; ++it) {
    const int p = it & 1;
    store_stage(p);
    __syncthreads();
    if (it + 1 < nT) load_stage((it + 1) * BK);   // overlap with WMMAs below

    // A fragments: lanes 0-15 hold K=0..15; lanes 16-31 hold K=16..31
    v16bf a0, a1, b0, b1;
    const int r0 = (wm * 32 + lr) * LDP;
    const int r1 = (wm * 32 + 16 + lr) * LDP;
#pragma unroll
    for (int j = 0; j < 8; ++j) {
      a0[j]     = As[p][r0 + khalf * 16 + j];
      a0[8 + j] = As[p][r0 + khalf * 16 + 8 + j];
      a1[j]     = As[p][r1 + khalf * 16 + j];
      a1[8 + j] = As[p][r1 + khalf * 16 + 8 + j];
    }
    // B fragments: lanes 0-15 hold K=0..15; lanes 16-31 hold K=16..31
#pragma unroll
    for (int j = 0; j < 16; ++j) {
      b0[j] = Ws[p][(wn * 32 + lr)      * LDP + khalf * 16 + j];
      b1[j] = Ws[p][(wn * 32 + 16 + lr) * LDP + khalf * 16 + j];
    }
    acc00 = WMMA_BF16(a0, b0, acc00);
    acc01 = WMMA_BF16(a0, b1, acc01);
    acc10 = WMMA_BF16(a1, b0, acc10);
    acc11 = WMMA_BF16(a1, b1, acc11);
  }

  const int m0 = blockIdx.y * BM + wm * 32;
  const int n0 = blockIdx.x * BN + wn * 32;
  const float bi0 = bias ? bias[n0 + lr]      : 0.f;
  const float bi1 = bias ? bias[n0 + 16 + lr] : 0.f;
#pragma unroll
  for (int r = 0; r < 8; ++r) {
    const int ma = m0 + khalf * 8 + r;        // C layout: VGPR r -> M=(lane>=16?8:0)+r
    const int mb = ma + 16;
    if (ma < M) {
      float c0 = acc00[r] + bi0, c1 = acc01[r] + bi1;
      if (RELU) { c0 = fmaxf(c0, 0.f); c1 = fmaxf(c1, 0.f); }
      const size_t i0 = (size_t)ma * N + n0 + lr;
      if (OUT_BF16) { Cb[i0] = (__bf16)c0; Cb[i0 + 16] = (__bf16)c1; }
      else          { Cf[i0] = c0;         Cf[i0 + 16] = c1; }
    }
    if (mb < M) {
      float c0 = acc10[r] + bi0, c1 = acc11[r] + bi1;
      if (RELU) { c0 = fmaxf(c0, 0.f); c1 = fmaxf(c1, 0.f); }
      const size_t i0 = (size_t)mb * N + n0 + lr;
      if (OUT_BF16) { Cb[i0] = (__bf16)c0; Cb[i0 + 16] = (__bf16)c1; }
      else          { Cf[i0] = c0;         Cf[i0 + 16] = c1; }
    }
  }
}

// ---------------------------------------------------------------------------
// t = LayerNorm(2*y) over (L, D)  -> f32 (batch-invariant, computed once)
// ---------------------------------------------------------------------------
__global__ __launch_bounds__(256)
void ln_double_kernel(const float* __restrict__ y, const float* __restrict__ g,
                      const float* __restrict__ bt, float* __restrict__ t)
{
  __shared__ float r1[256], r2[256];
  const int l = blockIdx.x, tid = threadIdx.x;
  float z[3]; float s = 0.f, sq = 0.f;
#pragma unroll
  for (int i = 0; i < 3; ++i) {
    z[i] = 2.f * y[(size_t)l * Dd + tid * 3 + i];
    s += z[i]; sq += z[i] * z[i];
  }
  r1[tid] = s; r2[tid] = sq; __syncthreads();
  for (int off = 128; off; off >>= 1) {
    if (tid < off) { r1[tid] += r1[tid + off]; r2[tid] += r2[tid + off]; }
    __syncthreads();
  }
  const float mean = r1[0] * (1.f / Dd);
  const float var  = r2[0] * (1.f / Dd) - mean * mean;
  const float inv  = rsqrtf(var + 1e-5f);
#pragma unroll
  for (int i = 0; i < 3; ++i) {
    const int d = tid * 3 + i;
    t[(size_t)l * Dd + d] = (z[i] - mean) * inv * g[d] + bt[d];
  }
}

// ---------------------------------------------------------------------------
// out = LayerNorm(Xa[row % aRows] + Xb[row])  -> f32 (+ optional bf16 copy)
// ---------------------------------------------------------------------------
__global__ __launch_bounds__(256)
void add_ln_kernel(const float* __restrict__ Xa, int aRows,
                   const float* __restrict__ Xb, const float* __restrict__ g,
                   const float* __restrict__ bt, float* __restrict__ outf,
                   __bf16* __restrict__ outb)
{
  __shared__ float r1[256], r2[256];
  const int row = blockIdx.x, tid = threadIdx.x;
  const int ar = row % aRows;
  float z[3]; float s = 0.f, sq = 0.f;
#pragma unroll
  for (int i = 0; i < 3; ++i) {
    const int d = tid * 3 + i;
    const float v = Xa[(size_t)ar * Dd + d] + Xb[(size_t)row * Dd + d];
    z[i] = v; s += v; sq += v * v;
  }
  r1[tid] = s; r2[tid] = sq; __syncthreads();
  for (int off = 128; off; off >>= 1) {
    if (tid < off) { r1[tid] += r1[tid + off]; r2[tid] += r2[tid + off]; }
    __syncthreads();
  }
  const float mean = r1[0] * (1.f / Dd);
  const float var  = r2[0] * (1.f / Dd) - mean * mean;
  const float inv  = rsqrtf(var + 1e-5f);
#pragma unroll
  for (int i = 0; i < 3; ++i) {
    const int d = tid * 3 + i;
    const float o = (z[i] - mean) * inv * g[d] + bt[d];
    outf[(size_t)row * Dd + d] = o;
    if (outb) outb[(size_t)row * Dd + d] = (__bf16)o;
  }
}

// ---------------------------------------------------------------------------
// Fused cross-attention: one workgroup per (batch, 16-row L-tile), loops heads.
// scores(16x1024 f32) + attn(16x1024 bf16) + head-avg accumulator(16x1024 f32)
// all live in LDS (~162 KB of the WGP's 320 KB). Writes ao as bf16 and
// a_score directly into d_out.
// ---------------------------------------------------------------------------
__global__ __launch_bounds__(256)
void attn_kernel(const __bf16* __restrict__ Qb, const __bf16* __restrict__ Kb,
                 const __bf16* __restrict__ Vb, __bf16* __restrict__ aob,
                 float* __restrict__ ascore)
{
  extern __shared__ char smem[];
  float*  sc    = (float*)smem;                 // 16*1024 f32   (64 KB)
  float*  asacc = sc + 16 * 1024;               // 16*1024 f32   (64 KB)
  __bf16* at    = (__bf16*)(asacc + 16 * 1024); // 16*1024 bf16  (32 KB)
  float*  red   = (float*)(at + 16 * 1024);     // 512 f32       ( 2 KB)

  const int tid = threadIdx.x, wv = tid >> 5, lane = tid & 31;
  const int lr = lane & 15, khalf = lane >> 4;
  const int b = blockIdx.y, l0 = blockIdx.x * 16;
  const float scale = 0.1020620726f; // 1/sqrt(96)

  for (int i = tid; i < 16 * 1024; i += 256) asacc[i] = 0.f;
  __syncthreads();

  for (int h = 0; h < Hh; ++h) {
    // ---- Q fragments (16 x 96 = 3 K-steps), same for every wave
    int ql = l0 + lr; if (ql > Ll - 1) ql = Ll - 1;
    v16bf qf[3];
#pragma unroll
    for (int kk = 0; kk < 3; ++kk) {
      const __bf16* qp = Qb + (size_t)ql * Dd + h * HDh + kk * 32;
#pragma unroll
      for (int j = 0; j < 8; ++j) {
        qf[kk][j]     = qp[khalf * 16 + j];
        qf[kk][8 + j] = qp[khalf * 16 + 8 + j];
      }
    }
    // ---- scores: 64 s-tiles of 16, 8 per wave
    for (int st = wv * 8; st < wv * 8 + 8; ++st) {
      const int s0 = st * 16;
      v8f acc = {};
#pragma unroll
      for (int kk = 0; kk < 3; ++kk) {
        const __bf16* kp = Kb + ((size_t)(b * Ss + s0 + lr)) * Dd
                              + h * HDh + kk * 32 + khalf * 16;
        v16bf bf;
#pragma unroll
        for (int j = 0; j < 16; ++j) bf[j] = kp[j];
        acc = WMMA_BF16(qf[kk], bf, acc);
      }
      const int mseg = khalf * 8;
#pragma unroll
      for (int r = 0; r < 8; ++r)
        sc[(mseg + r) * 1024 + s0 + lr] = acc[r] * scale;
    }
    __syncthreads();

    // ---- softmax: 16 threads per row, 64 cols each
    {
      const int row = tid >> 4, sub = tid & 15;
      float* srow = sc + row * 1024 + sub * 64;
      float mx = -1e30f;
      for (int j = 0; j < 64; ++j) mx = fmaxf(mx, srow[j]);
      red[row * 16 + sub] = mx;
      __syncthreads();
      float rm = -1e30f;
      for (int i = 0; i < 16; ++i) rm = fmaxf(rm, red[row * 16 + i]);
      float sum = 0.f;
      for (int j = 0; j < 64; ++j) { const float e = __expf(srow[j] - rm); srow[j] = e; sum += e; }
      red[256 + row * 16 + sub] = sum;
      __syncthreads();
      float rs = 0.f;
      for (int i = 0; i < 16; ++i) rs += red[256 + row * 16 + i];
      const float inv = 1.f / rs;
      __bf16* arow = at + row * 1024 + sub * 64;
      float*  aac  = asacc + row * 1024 + sub * 64;
      for (int j = 0; j < 64; ++j) {
        const float p = srow[j] * inv;
        arow[j] = (__bf16)p;
        aac[j] += 0.125f * p;   // mean over 8 heads
      }
    }
    __syncthreads();

    // ---- ao tile: 6 N-tiles of 16 over hd=96; waves 0..5 (wave-uniform EXEC)
    if (wv < 6) {
      const int ecol = h * HDh + wv * 16 + lr;
      v8f acc = {};
      for (int k0 = 0; k0 < Ss; k0 += 32) {
        v16bf af, bf;
#pragma unroll
        for (int j = 0; j < 8; ++j) {
          af[j]     = at[lr * 1024 + k0 + khalf * 16 + j];
          af[8 + j] = at[lr * 1024 + k0 + khalf * 16 + 8 + j];
        }
        const int kb = k0 + khalf * 16;
#pragma unroll
        for (int j = 0; j < 16; ++j)
          bf[j] = Vb[((size_t)(b * Ss + kb + j)) * Dd + ecol];
        acc = WMMA_BF16(af, bf, acc);
      }
      const int mseg = khalf * 8;
#pragma unroll
      for (int r = 0; r < 8; ++r) {
        const int l = l0 + mseg + r;
        if (l < Ll)
          aob[((size_t)(b * Ll + l)) * Dd + ecol] = (__bf16)acc[r];
      }
    }
    __syncthreads();
  }

  // ---- write head-averaged attention (a_score) straight into d_out
  for (int i = tid; i < 16 * 1024; i += 256) {
    const int m = i >> 10, s = i & 1023;
    const int l = l0 + m;
    if (l < Ll)
      ascore[((size_t)(b * Ll + l)) * Ss + s] = asacc[i];
  }
}

// ---------------------------------------------------------------------------
// GroupFC + flatten + bias: logits[b,c] (tiny: 0.1 GFLOP, scalar f32)
// ---------------------------------------------------------------------------
__global__ __launch_bounds__(256)
void groupfc_kernel(const float* __restrict__ hbuf, const float* __restrict__ y,
                    const float* __restrict__ pool, const float* __restrict__ dbias,
                    float* __restrict__ logits)
{
  const int idx = blockIdx.x * 256 + threadIdx.x;
  if (idx >= Bb * Cc) return;
  const int b = idx / Cc, c = idx % Cc;
  const int l = c / DFf, f = c % DFf;
  const float* hp = hbuf + (size_t)(b * Ll + l) * Dd;
  const float* yp = y + (size_t)l * Dd;
  const float* p0 = pool + (size_t)l * (2 * Dd) * DFf + f;
  float sum = dbias[c];
  for (int d = 0; d < Dd; ++d) sum += hp[d] * p0[d * DFf];
  const float* p1 = p0 + Dd * DFf;
  for (int d = 0; d < Dd; ++d) sum += yp[d] * p1[d * DFf];
  logits[idx] = sum;
}

// ---------------------------------------------------------------------------
extern "C" void kernel_launch(void* const* d_in, const int* in_sizes, int n_in,
                              void* d_out, int out_size, void* d_ws, size_t ws_size,
                              hipStream_t stream)
{
  const float* x    = (const float*)d_in[0];
  const float* y    = (const float*)d_in[1];
  const float* eW   = (const float*)d_in[2];
  const float* eb   = (const float*)d_in[3];
  const float* Wq   = (const float*)d_in[4];
  const float* Wk   = (const float*)d_in[5];
  const float* Wv   = (const float*)d_in[6];
  const float* bq   = (const float*)d_in[7];
  const float* bk   = (const float*)d_in[8];
  const float* bv   = (const float*)d_in[9];
  const float* Wo   = (const float*)d_in[10];
  const float* bo   = (const float*)d_in[11];
  const float* ln1g = (const float*)d_in[12];
  const float* ln1b = (const float*)d_in[13];
  const float* ln2g = (const float*)d_in[14];
  const float* ln2b = (const float*)d_in[15];
  const float* ln3g = (const float*)d_in[16];
  const float* ln3b = (const float*)d_in[17];
  const float* W1   = (const float*)d_in[18];
  const float* b1   = (const float*)d_in[19];
  const float* W2   = (const float*)d_in[20];
  const float* b2   = (const float*)d_in[21];
  const float* pool = (const float*)d_in[22];
  const float* dbias= (const float*)d_in[23];

  float* logits = (float*)d_out;
  float* ascore = logits + Bb * Cc;

  // workspace carve (256-byte aligned slices)
  char* wsb = (char*)d_ws;
  size_t off = 0;
  auto carve = [&](size_t bytes) -> char* {
    char* p = wsb + off;
    off += (bytes + 255) & ~(size_t)255;
    return p;
  };
  const size_t BS = (size_t)Bb * Ss;
  __bf16* mem_bf = (__bf16*)carve(BS * Dd * sizeof(__bf16));
  __bf16* K_bf   = (__bf16*)carve(BS * Dd * sizeof(__bf16));
  __bf16* V_bf   = (__bf16*)carve(BS * Dd * sizeof(__bf16));
  __bf16* q_bf   = (__bf16*)carve((size_t)Ll * Dd * sizeof(__bf16));
  float*  t_f    = (float*) carve((size_t)Ll * Dd * sizeof(float));
  __bf16* ao_bf  = (__bf16*)carve((size_t)Bb * Ll * Dd * sizeof(__bf16));
  float*  aoWo_f = (float*) carve((size_t)Bb * Ll * Dd * sizeof(float));
  float*  t2_f   = (float*) carve((size_t)Bb * Ll * Dd * sizeof(float));
  __bf16* t2_bf  = (__bf16*)carve((size_t)Bb * Ll * Dd * sizeof(__bf16));
  __bf16* f1_bf  = (__bf16*)carve((size_t)Bb * Ll * Dd * sizeof(__bf16));
  float*  f2_f   = (float*) carve((size_t)Bb * Ll * Dd * sizeof(float));
  float*  h_f    = (float*) carve((size_t)Bb * Ll * Dd * sizeof(float));
  (void)ws_size; (void)in_sizes; (void)n_in; (void)out_size;

  // 1) t = LN(2y), batch-invariant
  ln_double_kernel<<<dim3(Ll), dim3(256), 0, stream>>>(y, ln1g, ln1b, t_f);

  // 2) mem = relu(x @ eW^T + eb)  [M=32768, N=768, K=2048]  -> bf16
  gemm_bf16_wmma<true, true, true><<<dim3(Dd / 64, (Bb * Ss) / 128), dim3(256), 0, stream>>>(
      x, eW, eb, nullptr, mem_bf, Bb * Ss, Dd, F0);

  // 3/4) K = mem@Wk^T+bk, V = mem@Wv^T+bv  -> bf16
  gemm_bf16_wmma<false, false, true><<<dim3(Dd / 64, (Bb * Ss) / 128), dim3(256), 0, stream>>>(
      mem_bf, Wk, bk, nullptr, K_bf, Bb * Ss, Dd, Dd);
  gemm_bf16_wmma<false, false, true><<<dim3(Dd / 64, (Bb * Ss) / 128), dim3(256), 0, stream>>>(
      mem_bf, Wv, bv, nullptr, V_bf, Bb * Ss, Dd, Dd);

  // 5) Q = t@Wq^T+bq (batch-invariant, M=100) -> bf16
  gemm_bf16_wmma<true, false, true><<<dim3(Dd / 64, 1), dim3(256), 0, stream>>>(
      t_f, Wq, bq, nullptr, q_bf, Ll, Dd, Dd);

  // 6) fused attention: ao (bf16) + a_score -> d_out
  const size_t attn_lds = 16 * 1024 * sizeof(float) * 2   // sc + asacc
                        + 16 * 1024 * sizeof(__bf16)      // at
                        + 512 * sizeof(float);            // red
  attn_kernel<<<dim3(7, Bb), dim3(256), attn_lds, stream>>>(q_bf, K_bf, V_bf, ao_bf, ascore);

  // 7) aoWo = ao@Wo^T+bo  [M=3200] -> f32
  gemm_bf16_wmma<false, false, false><<<dim3(Dd / 64, (Bb * Ll) / 128), dim3(256), 0, stream>>>(
      ao_bf, Wo, bo, aoWo_f, nullptr, Bb * Ll, Dd, Dd);

  // 8) t2 = LN(t[l] + aoWo)  (t broadcast over batch)
  add_ln_kernel<<<dim3(Bb * Ll), dim3(256), 0, stream>>>(t_f, Ll, aoWo_f, ln2g, ln2b, t2_f, t2_bf);

  // 9) ffn1 = relu(t2@W1^T+b1) -> bf16 ; 10) ffn2 = ffn1@W2^T+b2 -> f32
  gemm_bf16_wmma<false, true, true><<<dim3(Dd / 64, (Bb * Ll) / 128), dim3(256), 0, stream>>>(
      t2_bf, W1, b1, nullptr, f1_bf, Bb * Ll, Dd, Dd);
  gemm_bf16_wmma<false, false, false><<<dim3(Dd / 64, (Bb * Ll) / 128), dim3(256), 0, stream>>>(
      f1_bf, W2, b2, f2_f, nullptr, Bb * Ll, Dd, Dd);

  // 11) h = LN(t2 + ffn2)
  add_ln_kernel<<<dim3(Bb * Ll), dim3(256), 0, stream>>>(t2_f, Bb * Ll, f2_f, ln3g, ln3b, h_f, nullptr);

  // 12) GroupFC + bias -> logits
  groupfc_kernel<<<dim3((Bb * Cc + 255) / 256), dim3(256), 0, stream>>>(h_f, y, pool, dbias, logits);
}
